// Operator_56573309224589
// MI455X (gfx1250) — matrix operand order repaired and checked by reading, not compiled, not run on hardware
//
#include <hip/hip_runtime.h>

typedef __attribute__((ext_vector_type(16))) _Float16 v16h;
typedef __attribute__((ext_vector_type(8)))  float    v8f;

#define HH     360
#define WW     720
#define BB     4
#define CC     2
#define KS     21
#define PADL   10
#define HP     (HH + 2*PADL)     // 380
#define WP     (WW + 2*PADL)     // 740
#define NOUT   24
#define NPADN  32                // NOUT padded to 2x16 (zero rows)
#define KCH    (CC*KS*KS)        // 882
#define KPADK  896               // 882 padded to 28*32
#define NKS    (KPADK/32)        // 28 k-steps
#define CNNP   (NOUT*KCH)        // 21168
#define PE     100
#define EMBD   40
#define MT     48                // x columns per block (3 M-subtiles of 16)
#define TW     (MT + KS - 1)     // 68 compact tile width

static __device__ __forceinline__ unsigned short h_bits(_Float16 h) {
    union { _Float16 h; unsigned short u; } cv; cv.h = h; return cv.u;
}

// ---------------------------------------------------------------------------
// K0: build zero-lat-padded / circular-lon-padded f16 image (B,C,380,740)
// ---------------------------------------------------------------------------
__global__ void k_pad_f16(const float* __restrict__ x, _Float16* __restrict__ xh) {
    int idx = blockIdx.x * 256 + threadIdx.x;
    const int total = BB * CC * HP * WP;
    if (idx >= total) return;
    int col = idx % WP;
    int t   = idx / WP;
    int row = t % HP;
    int bc  = t / HP;               // b*C + c
    int rs = row - PADL;
    int cs = col - PADL;
    if (cs < 0)   cs += WW;
    if (cs >= WW) cs -= WW;
    float v = 0.0f;
    if (rs >= 0 && rs < HH) v = x[((size_t)bc * HH + rs) * WW + cs];
    xh[idx] = (_Float16)v;
}

// ---------------------------------------------------------------------------
// K1: base[k] = b1[k] + sum_i cnn_kernel[i] * w1[i,k]   (shared by all rows)
// ---------------------------------------------------------------------------
__global__ void k_base(const float* __restrict__ ck, const float* __restrict__ w1,
                       const float* __restrict__ b1, float* __restrict__ base) {
    __shared__ float red[256];
    int k = blockIdx.x;             // 0..99
    float s = 0.0f;
    for (int i = threadIdx.x; i < CNNP; i += 256)
        s += ck[i] * w1[(size_t)i * PE + k];
    red[threadIdx.x] = s;
    __syncthreads();
    for (int off = 128; off > 0; off >>= 1) {
        if (threadIdx.x < off) red[threadIdx.x] += red[threadIdx.x + off];
        __syncthreads();
    }
    if (threadIdx.x == 0) base[k] = red[0] + b1[k];
}

// ---------------------------------------------------------------------------
// K2: per-row z1 = relu(base + pos_j @ w1_tail), z2 = relu(z1 @ w2 + b2)
// ---------------------------------------------------------------------------
__global__ void k_z2(const float* __restrict__ pos, const float* __restrict__ w1,
                     const float* __restrict__ base, const float* __restrict__ w2,
                     const float* __restrict__ b2, float* __restrict__ z2) {
    __shared__ float z1s[PE];
    int j = blockIdx.x;
    int k = threadIdx.x;
    if (k < PE) {
        float s = base[k];
        for (int t = 0; t < 4; ++t)
            s += pos[j * 4 + t] * w1[(size_t)(CNNP + t) * PE + k];
        z1s[k] = fmaxf(s, 0.0f);
    }
    __syncthreads();
    if (k < PE) {
        float s = b2[k];
        for (int l = 0; l < PE; ++l)
            s += z1s[l] * w2[l * PE + k];
        z2[j * PE + k] = fmaxf(s, 0.0f);
    }
}

// ---------------------------------------------------------------------------
// K3: kernh[j][n][kpad] (f16), n padded to 32 rows, K padded to 896: zeros
//     outside n<24 / k<882; else b3[p]+ck[p]+z2_j@w3[:,p], p=n*882+k
// ---------------------------------------------------------------------------
__global__ void k_kern(const float* __restrict__ z2, const float* __restrict__ w3,
                       const float* __restrict__ b3, const float* __restrict__ ck,
                       _Float16* __restrict__ kernh) {
    __shared__ float zs[PE];
    int j = blockIdx.x;
    if (threadIdx.x < PE) zs[threadIdx.x] = z2[j * PE + threadIdx.x];
    __syncthreads();
    for (int q = threadIdx.x; q < NPADN * KPADK; q += 256) {
        int k = q % KPADK;
        int n = q / KPADK;
        float v = 0.0f;
        if (k < KCH && n < NOUT) {
            int p = n * KCH + k;
            v = b3[p] + ck[p];
            for (int l = 0; l < PE; ++l)
                v += zs[l] * w3[(size_t)l * CNNP + p];
        }
        kernh[(size_t)j * NPADN * KPADK + q] = (_Float16)v;
    }
}

// ---------------------------------------------------------------------------
// K4: per-row conv as WMMA GEMM (M=48 tile, N=24->32, K=882->896) fused with
//     the pixelwise eval MLP. 6 waves/block: wave = (msub 0..2) x (nsub 0..1).
//     A staged in LDS in exact fragment order (pair-packed 32-bit stores) so
//     the inner loop is pure: ds_load_b128 x2 + global_load_b128 x2 + wmma.
// ---------------------------------------------------------------------------
__global__ void __launch_bounds__(192)
k_conv(const _Float16* __restrict__ xh, const _Float16* __restrict__ kernh,
       const float* __restrict__ ew1, const float* __restrict__ eb1,
       const float* __restrict__ ew2, const float* __restrict__ eb2,
       float* __restrict__ out) {
    __shared__ __attribute__((aligned(32))) unsigned int AfragU[3 * NKS * 256]; // 84KB
    __shared__ _Float16 tile[CC * KS * TW];                                     // 5.7KB
    __shared__ int      tileoff[KPADK];
    __shared__ float    feats[MT][NOUT + 1];

    int x0 = blockIdx.x * MT;
    int j  = blockIdx.y;
    int b  = blockIdx.z;
    int tid  = threadIdx.x;
    int lane = tid & 31;
    int wv   = tid >> 5;            // 0..5
    int msub = wv >> 1;             // 0..2
    int nsub = wv & 1;              // 0..1
    const _Float16* xbase = xh + (size_t)(b * CC) * HP * WP;

    // ---- stage 1: K->tile-offset table (decompose once per k) ----
    for (int k = tid; k < KPADK; k += 192) {
        int off = -1;
        if (k < KCH) {
            int cch = k / (KS * KS);
            int rem = k - cch * (KS * KS);
            int ky  = rem / KS;
            int kx  = rem - ky * KS;
            off = (cch * KS + ky) * TW + kx;
        }
        tileoff[k] = off;
    }
    // ---- stage 1b: compact patch tile, coalesced global loads ----
    for (int q = tid; q < CC * KS * TW; q += 192) {
        int row = q / TW;           // c*21 + ky
        int col = q - row * TW;
        int cch = row / KS;
        int ky  = row - cch * KS;
        tile[q] = xbase[((size_t)cch * HP + (j + ky)) * WP + (x0 + col)];
    }
    __syncthreads();

    // ---- stage 2: expand tile into fragment-order A, 2 halves per store ----
    // A 16x32 fragment order used here (same as B): lane<16: M=lane, K in {0..15};
    //                          lane>=16: M=lane-16, K in {16..31}.
    // Elements (e, e+1), e even, always map to K, K+1 -> pack into one u32.
    for (int q2 = tid; q2 < 3 * NKS * 256; q2 += 192) {
        int e2  = q2 & 7;           // pair index -> e = 2*e2
        int ln  = (q2 >> 3) & 31;
        int t2  = q2 >> 8;          // msub*NKS + kstep
        int ms  = t2 / NKS;
        int kst = t2 - ms * NKS;
        int e   = e2 << 1;
        int eK  = e + ((ln < 16) ? 0 : 16);
        int k   = kst * 32 + eK;
        int m   = ms * 16 + (ln & 15);
        int tof0 = tileoff[k];
        int tof1 = tileoff[k + 1];
        unsigned int lo = (tof0 >= 0) ? h_bits(tile[tof0 + m]) : 0u;
        unsigned int hi = (tof1 >= 0) ? h_bits(tile[tof1 + m]) : 0u;
        AfragU[q2] = lo | (hi << 16);
    }
    __syncthreads();

    // ---- stage 3: WMMA main loop: 1 LDS vec load + 1 global vec load + wmma
    int nloc   = (lane & 15) + nsub * 16;            // B column (zero-padded rows)
    int khalfB = (lane < 16) ? 0 : 16;
    const _Float16* brow =
        kernh + ((size_t)j * NPADN + nloc) * KPADK + khalfB;
    const _Float16* arow =
        (const _Float16*)&AfragU[(msub * NKS) * 256 + lane * 8];

    v8f acc = {};
#pragma unroll 4
    for (int ks = 0; ks < NKS; ++ks) {
        v16h a  = *(const v16h*)(arow + ks * 512);
        v16h bb = *(const v16h*)(brow + ks * 32);
        acc = __builtin_amdgcn_wmma_f32_16x16x32_f16(
            /*neg_a=*/false, a, /*neg_b=*/false, bb,
            /*c_mod=*/(short)0, acc, /*reuse_a=*/false, /*reuse_b=*/false);
    }

    // ---- scatter D (f32 16x16 layout: lane -> N=lane%16, vgpr r -> M=r(+8))
    {
        int Madd = (lane < 16) ? 0 : 8;
        int Nl   = lane & 15;
#pragma unroll
        for (int r = 0; r < 8; ++r) {
            int m = msub * 16 + r + Madd;
            int n = nsub * 16 + Nl;
            if (n < NOUT) feats[m][n] = acc[r];
        }
    }
    __syncthreads();

    // ---- fused eval MLP: relu(feat @ ew1 + eb1) @ ew2 + eb2 ----
    if (tid < MT) {
        float f[NOUT];
#pragma unroll
        for (int n = 0; n < NOUT; ++n) f[n] = feats[tid][n];
        float o0 = eb2[0], o1 = eb2[1];
        for (int u = 0; u < EMBD; ++u) {
            float h = eb1[u];
#pragma unroll
            for (int n = 0; n < NOUT; ++n) h += f[n] * ew1[n * EMBD + u];
            h = fmaxf(h, 0.0f);
            o0 += h * ew2[u * 2 + 0];
            o1 += h * ew2[u * 2 + 1];
        }
        int xg = x0 + tid;
        out[((size_t)(b * 2 + 0) * HH + j) * WW + xg] = o0;
        out[((size_t)(b * 2 + 1) * HH + j) * WW + xg] = o1;
    }
}

// ---------------------------------------------------------------------------
extern "C" void kernel_launch(void* const* d_in, const int* in_sizes, int n_in,
                              void* d_out, int out_size, void* d_ws, size_t ws_size,
                              hipStream_t stream) {
    (void)in_sizes; (void)n_in; (void)out_size; (void)ws_size;
    const float* x   = (const float*)d_in[0];
    const float* pos = (const float*)d_in[1];
    const float* ck  = (const float*)d_in[2];
    const float* w1  = (const float*)d_in[3];
    const float* b1  = (const float*)d_in[4];
    const float* w2  = (const float*)d_in[5];
    const float* b2  = (const float*)d_in[6];
    const float* w3  = (const float*)d_in[7];
    const float* b3  = (const float*)d_in[8];
    const float* ew1 = (const float*)d_in[9];
    const float* eb1 = (const float*)d_in[10];
    const float* ew2 = (const float*)d_in[11];
    const float* eb2 = (const float*)d_in[12];
    float* out = (float*)d_out;

    // Workspace layout (all offsets 256B-aligned)
    char* ws = (char*)d_ws;
    float*    base  = (float*)ws;                               //   400 B -> 512
    float*    z2    = (float*)(ws + 512);                       // 144000 B -> 144128
    _Float16* xh    = (_Float16*)(ws + 512 + 144128);           // 8*380*740*2 = 4499200 -> 4499456
    _Float16* kernh = (_Float16*)(ws + 512 + 144128 + 4499456); // 360*32*896*2 = 20643840

    const int padTotal = BB * CC * HP * WP;
    k_pad_f16<<<dim3((padTotal + 255) / 256), dim3(256), 0, stream>>>(x, xh);
    k_base   <<<dim3(PE),              dim3(256), 0, stream>>>(ck, w1, b1, base);
    k_z2     <<<dim3(HH),              dim3(128), 0, stream>>>(pos, w1, base, w2, b2, z2);
    k_kern   <<<dim3(HH),              dim3(256), 0, stream>>>(z2, w3, b3, ck, kernh);
    k_conv   <<<dim3(WW / MT, HH, BB), dim3(192), 0, stream>>>(xh, kernh, ew1, eb1, ew2, eb2, out);
}
